// HRLPolicy_53008486367770
// MI455X (gfx1250) — compile-verified, matrix operand order checked by reading, not run on hardware
//
#include <hip/hip_runtime.h>
#include <hip/hip_bf16.h>
#include <math.h>

#define NS 100000
#define NM 50000
#define NE 600000
#define DD 128
#define HORIZON 8
#define NEG_SLOPE 0.2f

typedef __attribute__((ext_vector_type(16))) __bf16 v16bf;
typedef __attribute__((ext_vector_type(8)))  float  v8f;

// ---------- helpers ----------
__device__ __forceinline__ unsigned fkey(float f) {
    unsigned b = __float_as_uint(f);
    return (b & 0x80000000u) ? ~b : (b | 0x80000000u);
}
__device__ __forceinline__ float fdec(unsigned k) {
    unsigned b = (k & 0x80000000u) ? (k & 0x7FFFFFFFu) : ~k;
    return __uint_as_float(b);
}
__device__ __forceinline__ unsigned pcg_hash(unsigned x) {
    x = x * 747796405u + 2891336453u;
    unsigned w = ((x >> ((x >> 28u) + 4u)) ^ x) * 277803737u;
    return (w >> 22u) ^ w;
}
__device__ __forceinline__ float sigmf(float x) { return 1.0f / (1.0f + __expf(-x)); }
__device__ __forceinline__ float wave_sum32(float v) {
    #pragma unroll
    for (int off = 16; off > 0; off >>= 1) v += __shfl_xor(v, off, 32);
    return v;
}

// ---------- init: zero accumulators every call ----------
__global__ void init_kernel(float* upd, float* segsum, unsigned* segmaxk,
                            float* g, float* h, float* c,
                            unsigned long long* argmax, unsigned* maxkey, float* sumexp) {
    int i = blockIdx.x * blockDim.x + threadIdx.x;
    if (i < NM * DD) upd[i] = 0.0f;
    if (i < NM) { segsum[i] = 0.0f; segmaxk[i] = 0u; }
    if (i < DD) { g[i] = 0.0f; h[i] = 0.0f; c[i] = 0.0f; }
    if (i == 0) { *argmax = 0ull; *maxkey = 0u; *sumexp = 0.0f; }
}

// ---------- pack W (row-major fp32 [128,128]) into bf16 WMMA B-fragments ----------
// fragment index: ((nt*4 + kt)*32 + lane)*16 + j
// lane 0-15: N = nt*16 + lane, K bases {0..7,16..23}; lane 16-31: K bases {8..15,24..31}
__global__ void prep_wfrag(const float* Wl, const float* Wr, __bf16* Fl, __bf16* Fr) {
    int i = blockIdx.x * blockDim.x + threadIdx.x;       // 0 .. 2*16384-1
    if (i >= 2 * 16384) return;
    int which = i >> 14;
    int r = i & 16383;
    int j    =  r        & 15;
    int lane = (r >> 4)  & 31;
    int kt   = (r >> 9)  & 3;
    int nt   =  r >> 11;
    int n  = nt * 16 + (lane & 15);
    int kb = kt * 32 + ((lane >> 4) ? 8 : 0);
    int k  = kb + (j < 8 ? j : 8 + j);                   // {kb..kb+7, kb+16..kb+23}
    const float* W = which ? Wr : Wl;
    __bf16*      F = which ? Fr : Fl;
    F[r] = (__bf16)W[n * DD + k];
}

// ---------- Y[rows,128] = X[rows,128] @ W^T via v_wmma_f32_16x16x32_bf16 ----------
// one wave per 16-row tile; K = 128 -> 4 wmma per N-tile, 8 N-tiles -> 32 wmma
__global__ __launch_bounds__(32) void wmma_gemm_xw(const float* __restrict__ X,
                                                   const __bf16* __restrict__ F,
                                                   float* __restrict__ Y) {
    int tile = blockIdx.x;
    int lane = threadIdx.x;
    int m  = tile * 16 + (lane & 15);
    int kb = (lane >> 4) * 8;
    const float* xrow = X + (size_t)m * DD;

    v16bf afr[4];
    #pragma unroll
    for (int kt = 0; kt < 4; ++kt) {
        int base = kt * 32 + kb;
        #pragma unroll
        for (int j = 0; j < 8; ++j) {
            afr[kt][j]     = (__bf16)xrow[base + j];
            afr[kt][8 + j] = (__bf16)xrow[base + 16 + j];
        }
    }
    const v16bf* BF = (const v16bf*)F;
    #pragma unroll
    for (int nt = 0; nt < 8; ++nt) {
        v8f acc = {};
        #pragma unroll
        for (int kt = 0; kt < 4; ++kt) {
            v16bf b = BF[(nt * 4 + kt) * 32 + lane];
            acc = __builtin_amdgcn_wmma_f32_16x16x32_bf16(
                false, afr[kt], false, b, (short)0, acc, false, false);
        }
        // C layout: lanes 0-15 -> N=lane, M=r; lanes 16-31 -> N=lane-16, M=r+8
        int n    = nt * 16 + (lane & 15);
        int mrow = tile * 16 + (lane >> 4) * 8;
        #pragma unroll
        for (int r = 0; r < 8; ++r)
            Y[(size_t)(mrow + r) * DD + n] = acc[r];
    }
}

// ---------- edge score: e = att . leaky(x_l[src] + x_r[dst]); atomic segmax ----------
__global__ __launch_bounds__(256) void edge_score(const int* __restrict__ ei,
                                                  const float* __restrict__ xl,
                                                  const float* __restrict__ xr,
                                                  const float* __restrict__ att,
                                                  float* __restrict__ e,
                                                  unsigned* __restrict__ segmaxk) {
    int gw   = (blockIdx.x * blockDim.x + threadIdx.x) >> 5;  // wave per edge
    int lane = threadIdx.x & 31;
    int s = ei[gw];
    int d = ei[NE + gw];
    float4 a  = ((const float4*)(xl + (size_t)s * DD))[lane];
    float4 b  = ((const float4*)(xr + (size_t)d * DD))[lane];
    float4 at = ((const float4*)att)[lane];
    float zx = a.x + b.x, zy = a.y + b.y, zz = a.z + b.z, zw = a.w + b.w;
    zx = zx > 0.f ? zx : NEG_SLOPE * zx;
    zy = zy > 0.f ? zy : NEG_SLOPE * zy;
    zz = zz > 0.f ? zz : NEG_SLOPE * zz;
    zw = zw > 0.f ? zw : NEG_SLOPE * zw;
    float acc = wave_sum32(zx * at.x + zy * at.y + zz * at.z + zw * at.w);
    if (lane == 0) {
        e[gw] = acc;
        atomicMax(&segmaxk[d], fkey(acc));
    }
}

// ---------- exp(e - max[dst]) and segment sum ----------
__global__ void edge_exp(const int* __restrict__ ei, float* __restrict__ e,
                         const unsigned* __restrict__ segmaxk, float* __restrict__ segsum) {
    int i = blockIdx.x * blockDim.x + threadIdx.x;
    if (i >= NE) return;
    int d = ei[NE + i];
    float ex = __expf(e[i] - fdec(segmaxk[d]));
    e[i] = ex;
    atomicAdd(&segsum[d], ex);
}

// ---------- upd[dst] += alpha * x_l[src] ----------
__global__ __launch_bounds__(256) void edge_aggregate(const int* __restrict__ ei,
                                                      const float* __restrict__ xl,
                                                      const float* __restrict__ eex,
                                                      const float* __restrict__ segsum,
                                                      float* __restrict__ upd) {
    int gw   = (blockIdx.x * blockDim.x + threadIdx.x) >> 5;
    int lane = threadIdx.x & 31;
    int s = ei[gw];
    int d = ei[NE + gw];
    float alpha = eex[gw] / segsum[d];
    float4 xv = ((const float4*)(xl + (size_t)s * DD))[lane];
    float* o = upd + (size_t)d * DD + lane * 4;
    atomicAdd(o + 0, alpha * xv.x);
    atomicAdd(o + 1, alpha * xv.y);
    atomicAdd(o + 2, alpha * xv.z);
    atomicAdd(o + 3, alpha * xv.w);
}

// ---------- add conv bias; g = column mean ----------
__global__ __launch_bounds__(128) void bias_mean(float* __restrict__ upd,
                                                 const float* __restrict__ bias,
                                                 float* __restrict__ g) {
    int d  = threadIdx.x;
    int r0 = blockIdx.x * 64;
    int r1 = r0 + 64; if (r1 > NM) r1 = NM;
    float bv = bias[d], bsum = 0.0f;
    for (int r = r0; r < r1; ++r) {
        float v = upd[(size_t)r * DD + d] + bv;
        upd[(size_t)r * DD + d] = v;
        bsum += v;
    }
    atomicAdd(&g[d], bsum * (1.0f / (float)NM));
}

// ---------- LSTM cell + predictor hidden layer (1 block, 128 threads) ----------
__global__ __launch_bounds__(128) void lstm_cell(const float* __restrict__ inp_src,
                                                 float* __restrict__ h, float* __restrict__ c,
                                                 float* __restrict__ hidden,
                                                 const float* __restrict__ Wih,
                                                 const float* __restrict__ Whh,
                                                 const float* __restrict__ bih,
                                                 const float* __restrict__ bhh,
                                                 const float* __restrict__ W1,
                                                 const float* __restrict__ b1) {
    __shared__ float sin[DD], sh[DD];
    int j = threadIdx.x;
    sin[j] = inp_src[j];
    sh[j]  = h[j];
    __syncthreads();
    float acc[4];
    #pragma unroll
    for (int gt = 0; gt < 4; ++gt) {
        int row = gt * DD + j;
        float a = bih[row] + bhh[row];
        const float* wi = Wih + (size_t)row * DD;
        const float* wh = Whh + (size_t)row * DD;
        for (int k = 0; k < DD; ++k) a = fmaf(wi[k], sin[k], fmaf(wh[k], sh[k], a));
        acc[gt] = a;
    }
    float iv = sigmf(acc[0]), fv = sigmf(acc[1]);
    float gv = tanhf(acc[2]),  ov = sigmf(acc[3]);
    float cn = fv * c[j] + iv * gv;
    float hn = ov * tanhf(cn);
    c[j] = cn; h[j] = hn;
    __syncthreads();
    sh[j] = hn;
    __syncthreads();
    float a = b1[j];
    const float* w1 = W1 + (size_t)j * DD;
    for (int k = 0; k < DD; ++k) a = fmaf(w1[k], sh[k], a);
    hidden[j] = fmaxf(a, 0.0f);
}

// ---------- logits GEMV + Gumbel argmax + running max (wave per row) ----------
__global__ __launch_bounds__(256) void logits_kernel(const float* __restrict__ W2,
                                                     const float* __restrict__ b2,
                                                     const float* __restrict__ hidden,
                                                     float* __restrict__ logits,
                                                     unsigned* __restrict__ maxkey,
                                                     unsigned long long* __restrict__ argmax,
                                                     int t) {
    __shared__ float hsh[DD];
    if (threadIdx.x < DD) hsh[threadIdx.x] = hidden[threadIdx.x];
    __syncthreads();
    int wave = threadIdx.x >> 5, lane = threadIdx.x & 31;
    int j = blockIdx.x * 8 + wave;                       // NM % 8 == 0
    float4 w = ((const float4*)(W2 + (size_t)j * DD))[lane];
    float acc = w.x * hsh[lane * 4] + w.y * hsh[lane * 4 + 1] +
                w.z * hsh[lane * 4 + 2] + w.w * hsh[lane * 4 + 3];
    acc = wave_sum32(acc);
    if (lane == 0) {
        float logit = acc + b2[j];
        logits[j] = logit;
        atomicMax(maxkey, fkey(logit));
        // deterministic Gumbel(0,1) keyed on (step, row) -- Gumbel-max categorical sample
        unsigned r = pcg_hash(((unsigned)j * 8u + (unsigned)t) ^ 0x2A2A9E37u);
        float u = (float)(r >> 8) * (1.0f / 16777216.0f) + 1e-10f;
        float gum = -__logf(-__logf(u));
        unsigned long long pk =
            ((unsigned long long)fkey(logit + gum) << 32) |
            (unsigned long long)(0x7FFFFFFFu - (unsigned)j);   // smallest index wins ties
        atomicMax(argmax, pk);
    }
}

// ---------- sum of exp(logit - max) ----------
__global__ __launch_bounds__(256) void sumexp_kernel(const float* __restrict__ logits,
                                                     const unsigned* __restrict__ maxkey,
                                                     float* __restrict__ sum) {
    __shared__ float sred[256];
    int i = blockIdx.x * 256 + threadIdx.x;
    float m = fdec(*maxkey);
    float v = (i < NM) ? __expf(logits[i] - m) : 0.0f;
    sred[threadIdx.x] = v;
    __syncthreads();
    for (int s = 128; s > 0; s >>= 1) {
        if (threadIdx.x < s) sred[threadIdx.x] += sred[threadIdx.x + s];
        __syncthreads();
    }
    if (threadIdx.x == 0) atomicAdd(sum, sred[0]);
}

// ---------- pick action, log_prob, feed embedding forward, reset scalars ----------
__global__ __launch_bounds__(128) void finalize_step(const float* __restrict__ logits,
                                                     unsigned long long* __restrict__ argmax,
                                                     unsigned* __restrict__ maxkey,
                                                     float* __restrict__ sumexp,
                                                     const float* __restrict__ emb,
                                                     float* __restrict__ inp,
                                                     float* __restrict__ out, int t) {
    __shared__ int sact;
    if (threadIdx.x == 0) {
        unsigned long long pk = *argmax;
        int action = (int)(0x7FFFFFFFu - (unsigned)(pk & 0xFFFFFFFFu));
        float maxl = fdec(*maxkey);
        float lse  = maxl + __logf(*sumexp);
        out[t]            = (float)action;
        out[HORIZON + t]  = logits[action] - lse;
        sact = action;
        *argmax = 0ull; *maxkey = 0u; *sumexp = 0.0f;     // reset for next step
    }
    __syncthreads();
    inp[threadIdx.x] = emb[(size_t)sact * DD + threadIdx.x];
}

__global__ __launch_bounds__(128) void copy_hc(const float* __restrict__ h,
                                               const float* __restrict__ c,
                                               float* __restrict__ out) {
    int i = threadIdx.x;
    out[2 * HORIZON + i]      = h[i];
    out[2 * HORIZON + DD + i] = c[i];
}

// ---------------- host ----------------
extern "C" void kernel_launch(void* const* d_in, const int* in_sizes, int n_in,
                              void* d_out, int out_size, void* d_ws, size_t ws_size,
                              hipStream_t stream) {
    const float* state_f  = (const float*)d_in[0];
    const float* model_f  = (const float*)d_in[1];
    const int*   ei       = (const int*)d_in[2];
    const float* W_l      = (const float*)d_in[3];
    const float* W_r      = (const float*)d_in[4];
    const float* att      = (const float*)d_in[5];
    const float* conv_b   = (const float*)d_in[6];
    const float* Wih      = (const float*)d_in[7];
    const float* Whh      = (const float*)d_in[8];
    const float* bih      = (const float*)d_in[9];
    const float* bhh      = (const float*)d_in[10];
    const float* pW1      = (const float*)d_in[11];
    const float* pb1      = (const float*)d_in[12];
    const float* pW2      = (const float*)d_in[13];
    const float* pb2      = (const float*)d_in[14];
    const float* emb      = (const float*)d_in[15];
    float* out = (float*)d_out;

    // workspace layout (256B aligned)
    char* ws = (char*)d_ws;
    size_t off = 0;
    auto alloc = [&](size_t bytes) { size_t r = off; off = (off + bytes + 255) & ~(size_t)255; return r; };
    float*  xl      = (float*)(ws + alloc((size_t)NS * DD * 4));
    float*  xr      = (float*)(ws + alloc((size_t)NM * DD * 4));
    float*  upd     = (float*)(ws + alloc((size_t)NM * DD * 4));
    float*  ebuf    = (float*)(ws + alloc((size_t)NE * 4));
    unsigned* segmk = (unsigned*)(ws + alloc((size_t)NM * 4));
    float*  segsum  = (float*)(ws + alloc((size_t)NM * 4));
    float*  logits  = (float*)(ws + alloc((size_t)NM * 4));
    __bf16* fragL   = (__bf16*)(ws + alloc(16384 * 2));
    __bf16* fragR   = (__bf16*)(ws + alloc(16384 * 2));
    float*  g       = (float*)(ws + alloc(DD * 4));
    float*  h       = (float*)(ws + alloc(DD * 4));
    float*  c       = (float*)(ws + alloc(DD * 4));
    float*  inp     = (float*)(ws + alloc(DD * 4));
    float*  hidden  = (float*)(ws + alloc(DD * 4));
    char*   scal    = ws + alloc(256);
    unsigned long long* argmax = (unsigned long long*)(scal + 0);
    unsigned*           maxkey = (unsigned*)(scal + 8);
    float*              sumexp = (float*)(scal + 12);
    (void)ws_size; (void)in_sizes; (void)n_in; (void)out_size;

    // 1) init accumulators
    init_kernel<<<(NM * DD + 255) / 256, 256, 0, stream>>>(
        upd, segsum, segmk, g, h, c, argmax, maxkey, sumexp);
    // 2) pack WMMA B fragments (bf16)
    prep_wfrag<<<(2 * 16384 + 255) / 256, 256, 0, stream>>>(W_l, W_r, fragL, fragR);
    // 3) x_l, x_r via bf16 WMMA
    wmma_gemm_xw<<<NS / 16, 32, 0, stream>>>(state_f, fragL, xl);
    wmma_gemm_xw<<<NM / 16, 32, 0, stream>>>(model_f, fragR, xr);
    // 4) edge phase: score -> softmax -> aggregate
    edge_score    <<<NE / 8, 256, 0, stream>>>(ei, xl, xr, att, ebuf, segmk);
    edge_exp      <<<(NE + 255) / 256, 256, 0, stream>>>(ei, ebuf, segmk, segsum);
    edge_aggregate<<<NE / 8, 256, 0, stream>>>(ei, xl, ebuf, segsum, upd);
    // 5) bias + global mean g
    bias_mean<<<(NM + 63) / 64, 128, 0, stream>>>(upd, conv_b, g);
    // 6) sequential rollout
    for (int t = 0; t < HORIZON; ++t) {
        const float* cur_inp = (t == 0) ? g : inp;
        lstm_cell<<<1, 128, 0, stream>>>(cur_inp, h, c, hidden,
                                         Wih, Whh, bih, bhh, pW1, pb1);
        logits_kernel<<<NM / 8, 256, 0, stream>>>(pW2, pb2, hidden, logits,
                                                  maxkey, argmax, t);
        sumexp_kernel<<<(NM + 255) / 256, 256, 0, stream>>>(logits, maxkey, sumexp);
        finalize_step<<<1, 128, 0, stream>>>(logits, argmax, maxkey, sumexp,
                                             emb, inp, out, t);
    }
    copy_hc<<<1, 128, 0, stream>>>(h, c, out);
}
